// DFlashAttention_63453846831162
// MI455X (gfx1250) — compile-verified, matrix operand order checked by reading, not run on hardware
//
#include <hip/hip_runtime.h>

#define DEVFN __device__ __forceinline__

constexpr int B_     = 2;
constexpr int CTX_   = 4096;
constexpr int DRAFT_ = 512;
constexpr int D_     = 2048;
constexpr int H_     = 16;
constexpr int KVH_   = 4;
constexpr int HD_    = 128;
constexpr int TOTAL_ = CTX_ + DRAFT_;     // 4608
constexpr float EPS_   = 1e-6f;
constexpr float SCALE_ = 0.08838834764831845f;   // 1/sqrt(128)
constexpr float LOG_THETA_OVER_HALF_ = 0.14391156616f;  // ln(10000)/64

typedef unsigned short u16;
typedef __attribute__((ext_vector_type(8)))  unsigned short u16x8;
typedef __attribute__((ext_vector_type(16))) __bf16         v16bf;
typedef __attribute__((ext_vector_type(8)))  float          v8f;

union FragU { v16bf v; u16x8 h[2]; };

DEVFN u16 f2bf(float f) {
  unsigned u = __float_as_uint(f);
  u += 0x7fffu + ((u >> 16) & 1u);           // round-to-nearest-even
  return (u16)(u >> 16);
}

DEVFN v16bf load_frag(const u16* p) {
  FragU f;
  f.h[0] = *(const u16x8*)(p);
  f.h[1] = *(const u16x8*)(p + 16);
  return f.v;
}

DEVFN v8f zero8() {
  v8f z;
#pragma unroll
  for (int i = 0; i < 8; ++i) z[i] = 0.0f;
  return z;
}

DEVFN v8f wmma_bf16(v16bf a, v16bf b, v8f c) {
  // D = A(16x32 bf16) * B(32x16 bf16) + C(16x16 f32)
  return __builtin_amdgcn_wmma_f32_16x16x32_bf16(false, a, false, b, (short)0, c,
                                                 false, false);
}

// WGP-scope prefetch (locality 3 -> near-cache fill), for streaming operands.
DEVFN void prefetch_near(const void* p) { __builtin_prefetch(p, 0, 3); }

// ---------------------------------------------------------------------------
// fp32 -> bf16 conversion with batched destination stride (for KV concat)
// ---------------------------------------------------------------------------
__global__ void cvt_bf16_kernel(const float* __restrict__ src, u16* __restrict__ dst,
                                int batches, int perBatch, long long dstStride) {
  long long n = (long long)batches * perBatch;
  for (long long i = (long long)blockIdx.x * blockDim.x + threadIdx.x; i < n;
       i += (long long)gridDim.x * blockDim.x) {
    long long b = i / perBatch;
    long long r = i - b * perBatch;
    dst[b * dstStride + r] = f2bf(src[i]);
  }
}

// ---------------------------------------------------------------------------
// Fused projection: C = X @ W^T, then per-head RMSNorm + RoPE (Q/K), stores
// bf16 in WMMA-friendly layouts.  mode: 0=Q, 1=K, 2=V (V stored transposed).
// One wave = 16 rows x 128 cols (exactly one head strip).
// ---------------------------------------------------------------------------
__global__ void __launch_bounds__(256)
proj_kernel(const u16* __restrict__ X,      // kvin  [B*TOTAL, D] bf16
            const u16* __restrict__ W,      // [N, D] bf16
            const float* __restrict__ normw,
            const int* __restrict__ ctx_pos, const int* __restrict__ draft_pos,
            u16* __restrict__ out, int mode) {
  const int lane = threadIdx.x & 31;
  const int wid  = blockIdx.x * (blockDim.x >> 5) + (threadIdx.x >> 5);
  const int half = lane >> 4;
  const int nl   = lane & 15;
  const int koff = half ? 8 : 0;

  int rt, head;
  if (mode == 0) { rt = wid & 63;                 head = wid >> 6; }
  else           { rt = wid % (B_ * TOTAL_ / 16); head = wid / (B_ * TOTAL_ / 16); }

  const int row0 = rt * 16;
  int b, s0;
  long long growA;
  if (mode == 0) { b = row0 / DRAFT_; s0 = row0 % DRAFT_;
                   growA = (long long)b * TOTAL_ + CTX_ + s0 + nl; }
  else           { b = row0 / TOTAL_; s0 = row0 % TOTAL_;
                   growA = (long long)row0 + nl; }

  const u16* arow = X + growA * D_ + koff;
  const u16* wptr[8];
#pragma unroll
  for (int t = 0; t < 8; ++t) {
    int col = head * HD_ + t * 16 + nl;
    wptr[t] = W + (long long)col * D_ + koff;
  }

  v8f acc[8];
#pragma unroll
  for (int t = 0; t < 8; ++t) acc[t] = zero8();

  for (int k0 = 0; k0 < D_; k0 += 32) {
    v16bf a = load_frag(arow + k0);
    prefetch_near(arow + k0 + 128);      // 4 K-steps ahead on streaming rows
#pragma unroll
    for (int t = 0; t < 8; ++t) {
      v16bf bf = load_frag(wptr[t] + k0);
      acc[t] = wmma_bf16(a, bf, acc[t]);
    }
  }

  if (mode == 2) {
    // V: store transposed -> vT[b][kvh][hd][s]
#pragma unroll
    for (int t = 0; t < 8; ++t)
#pragma unroll
      for (int r = 0; r < 8; ++r) {
        int m  = r + 8 * half;
        int hd = t * 16 + nl;
        long long idx = ((long long)(b * KVH_ + head) * HD_ + hd) * TOTAL_ + (s0 + m);
        out[idx] = f2bf(acc[t][r]);
      }
    return;
  }

  // ---- per-head RMSNorm over HD=128 (fp32) ----
  float ss[8];
#pragma unroll
  for (int r = 0; r < 8; ++r) {
    float s = 0.0f;
#pragma unroll
    for (int t = 0; t < 8; ++t) s += acc[t][r] * acc[t][r];
    ss[r] = s;
  }
#pragma unroll
  for (int off = 1; off < 16; off <<= 1)
#pragma unroll
    for (int r = 0; r < 8; ++r) ss[r] += __shfl_xor(ss[r], off, 32);
  float rs[8];
#pragma unroll
  for (int r = 0; r < 8; ++r) rs[r] = rsqrtf(ss[r] * (1.0f / HD_) + EPS_);

  float wn[8];
#pragma unroll
  for (int t = 0; t < 8; ++t) wn[t] = normw[t * 16 + nl];
#pragma unroll
  for (int t = 0; t < 8; ++t)
#pragma unroll
    for (int r = 0; r < 8; ++r) acc[t][r] = acc[t][r] * rs[r] * wn[t];

  // ---- RoPE: pairs are (d, d+64) i.e. tiles (t, t+4); cos/sin shared ----
  float pos[8];
#pragma unroll
  for (int r = 0; r < 8; ++r) {
    int m = r + 8 * half;
    int p;
    if (mode == 0) p = draft_pos[b * DRAFT_ + s0 + m];
    else {
      int s = s0 + m;
      p = (s < CTX_) ? ctx_pos[b * CTX_ + s] : draft_pos[b * DRAFT_ + (s - CTX_)];
    }
    pos[r] = (float)p;
  }
#pragma unroll
  for (int t = 0; t < 4; ++t) {
    float invf = __expf(-(float)(t * 16 + nl) * LOG_THETA_OVER_HALF_);
#pragma unroll
    for (int r = 0; r < 8; ++r) {
      float ang = pos[r] * invf;
      float c = __cosf(ang), s = __sinf(ang);
      float x1 = acc[t][r], x2 = acc[t + 4][r];
      acc[t][r]     = x1 * c - x2 * s;
      acc[t + 4][r] = x2 * c + x1 * s;
    }
  }

  // ---- store bf16: Q -> [b,h,q,hd], K -> [b,kvh,s,hd] (hd contiguous) ----
#pragma unroll
  for (int t = 0; t < 8; ++t)
#pragma unroll
    for (int r = 0; r < 8; ++r) {
      int m  = r + 8 * half;
      int hd = t * 16 + nl;
      long long idx;
      if (mode == 0) idx = ((long long)(b * H_   + head) * DRAFT_ + (s0 + m)) * HD_ + hd;
      else           idx = ((long long)(b * KVH_ + head) * TOTAL_ + (s0 + m)) * HD_ + hd;
      out[idx] = f2bf(acc[t][r]);
    }
}

// ---------------------------------------------------------------------------
// Flash attention: one wave per (b, h, 16-query tile); 8 waves/block grouped
// so all 4 heads of a KV head share K/V traffic.  32-key chunks; fp32 online
// softmax; P transposed via 1 KB LDS into bf16 A-fragments.
// ---------------------------------------------------------------------------
__global__ void __launch_bounds__(256)
attn_kernel(const u16* __restrict__ q_rope,   // [B,H,DRAFT,HD] bf16
            const u16* __restrict__ k_rope,   // [B,KVH,TOTAL,HD] bf16
            const u16* __restrict__ vT,       // [B,KVH,HD,TOTAL] bf16
            u16* __restrict__ attn) {         // [B,DRAFT,H*HD] bf16
  __shared__ __align__(16) u16 plds[8][16 * 32];

  const int lane = threadIdx.x & 31;
  const int w    = threadIdx.x >> 5;
  const int half = lane >> 4;
  const int nl   = lane & 15;
  const int koff = half ? 8 : 0;

  const int bid = blockIdx.x;                  // B*KVH*16 = 128 blocks
  const int b   = bid / (KVH_ * 16);
  const int kvh = (bid / 16) % KVH_;
  const int qp  = bid % 16;
  const int h   = kvh * 4 + (w & 3);
  const int q0  = (qp * 2 + (w >> 2)) * 16;

  // Q fragments (16 queries x 128 hd = 4 A-frags), loaded once
  v16bf qf[4];
  const u16* qbase = q_rope + ((long long)(b * H_ + h) * DRAFT_ + q0 + nl) * HD_ + koff;
#pragma unroll
  for (int kk = 0; kk < 4; ++kk) qf[kk] = load_frag(qbase + kk * 32);

  v8f o[8];
#pragma unroll
  for (int t = 0; t < 8; ++t) o[t] = zero8();
  float mst[8], lst[8];
#pragma unroll
  for (int r = 0; r < 8; ++r) { mst[r] = -1e30f; lst[r] = 0.0f; }

  const u16* kbase = k_rope + (long long)(b * KVH_ + kvh) * TOTAL_ * HD_;
  const u16* vbase = vT     + (long long)(b * KVH_ + kvh) * HD_ * TOTAL_;

  for (int s0c = 0; s0c < TOTAL_; s0c += 32) {
    // ---- S = Q @ K^T for 32 keys ----
    v8f st[2];
#pragma unroll
    for (int j = 0; j < 2; ++j) {
      st[j] = zero8();
      const u16* kp = kbase + (long long)(s0c + j * 16 + nl) * HD_ + koff;
      prefetch_near(kp + 32 * HD_);            // next key chunk
#pragma unroll
      for (int kk = 0; kk < 4; ++kk) {
        v16bf kf = load_frag(kp + kk * 32);
        st[j] = wmma_bf16(qf[kk], kf, st[j]);
      }
    }

    // ---- online softmax (fp32) ----
    float cm[8];
#pragma unroll
    for (int r = 0; r < 8; ++r) {
      float a = st[0][r] * SCALE_, c = st[1][r] * SCALE_;
      st[0][r] = a; st[1][r] = c;
      cm[r] = fmaxf(a, c);
    }
#pragma unroll
    for (int off = 1; off < 16; off <<= 1)
#pragma unroll
      for (int r = 0; r < 8; ++r) cm[r] = fmaxf(cm[r], __shfl_xor(cm[r], off, 32));

    float corr[8], cs[8];
#pragma unroll
    for (int r = 0; r < 8; ++r) {
      float mn = fmaxf(mst[r], cm[r]);
      corr[r] = __expf(mst[r] - mn);
      mst[r] = mn;
      float p0 = __expf(st[0][r] - mn);
      float p1 = __expf(st[1][r] - mn);
      st[0][r] = p0; st[1][r] = p1;
      cs[r] = p0 + p1;
    }
#pragma unroll
    for (int off = 1; off < 16; off <<= 1)
#pragma unroll
      for (int r = 0; r < 8; ++r) cs[r] += __shfl_xor(cs[r], off, 32);
#pragma unroll
    for (int r = 0; r < 8; ++r) lst[r] = lst[r] * corr[r] + cs[r];
#pragma unroll
    for (int t = 0; t < 8; ++t)
#pragma unroll
      for (int r = 0; r < 8; ++r) o[t][r] *= corr[r];

    // ---- P: C-layout -> LDS -> bf16 A-fragment (16x32) ----
#pragma unroll
    for (int j = 0; j < 2; ++j)
#pragma unroll
      for (int r = 0; r < 8; ++r)
        plds[w][(r + 8 * half) * 32 + j * 16 + nl] = f2bf(st[j][r]);

    v16bf pf = load_frag(&plds[w][nl * 32 + koff]);

    // ---- O += P @ V (V^T layout: contiguous keys per hd column) ----
#pragma unroll
    for (int t = 0; t < 8; ++t) {
      const u16* vp = vbase + (long long)(t * 16 + nl) * TOTAL_ + s0c + koff;
      prefetch_near(vp + 32);                  // next key chunk of V^T row
      v16bf vf = load_frag(vp);
      o[t] = wmma_bf16(pf, vf, o[t]);
    }
  }

  // ---- normalize and store attn bf16 [b, q, h*HD + hd] ----
#pragma unroll
  for (int r = 0; r < 8; ++r) lst[r] = 1.0f / lst[r];
#pragma unroll
  for (int t = 0; t < 8; ++t)
#pragma unroll
    for (int r = 0; r < 8; ++r) {
      int m = r + 8 * half;
      long long idx = ((long long)b * DRAFT_ + q0 + m) * (H_ * HD_) + h * HD_ + t * 16 + nl;
      attn[idx] = f2bf(o[t][r] * lst[r]);
    }
}

// ---------------------------------------------------------------------------
// Output projection: out(fp32) = attn(bf16) @ Wo(bf16)^T
// ---------------------------------------------------------------------------
__global__ void __launch_bounds__(256)
out_gemm_kernel(const u16* __restrict__ A, const u16* __restrict__ Wo,
                float* __restrict__ out) {
  const int lane = threadIdx.x & 31;
  const int wid  = blockIdx.x * (blockDim.x >> 5) + (threadIdx.x >> 5);
  const int half = lane >> 4;
  const int nl   = lane & 15;
  const int koff = half ? 8 : 0;

  const int rt     = wid & 63;   // 64 row tiles (B*DRAFT = 1024 rows)
  const int cstrip = wid >> 6;   // 16 strips of 128 cols
  const int row0   = rt * 16;

  const u16* arow = A + (long long)(row0 + nl) * (H_ * HD_) + koff;
  const u16* wptr[8];
#pragma unroll
  for (int t = 0; t < 8; ++t)
    wptr[t] = Wo + (long long)(cstrip * 128 + t * 16 + nl) * (H_ * HD_) + koff;

  v8f acc[8];
#pragma unroll
  for (int t = 0; t < 8; ++t) acc[t] = zero8();

  for (int k0 = 0; k0 < H_ * HD_; k0 += 32) {
    v16bf a = load_frag(arow + k0);
    prefetch_near(arow + k0 + 128);
#pragma unroll
    for (int t = 0; t < 8; ++t) {
      v16bf bf = load_frag(wptr[t] + k0);
      acc[t] = wmma_bf16(a, bf, acc[t]);
    }
  }

#pragma unroll
  for (int t = 0; t < 8; ++t)
#pragma unroll
    for (int r = 0; r < 8; ++r) {
      int m = r + 8 * half;
      out[(long long)(row0 + m) * D_ + cstrip * 128 + t * 16 + nl] = acc[t][r];
    }
}

// ---------------------------------------------------------------------------
extern "C" void kernel_launch(void* const* d_in, const int* in_sizes, int n_in,
                              void* d_out, int out_size, void* d_ws, size_t ws_size,
                              hipStream_t stream) {
  (void)in_sizes; (void)n_in; (void)out_size; (void)ws_size;

  const float* draft_hidden   = (const float*)d_in[0];
  const float* context_hidden = (const float*)d_in[1];
  const float* Wq             = (const float*)d_in[2];
  const float* Wk             = (const float*)d_in[3];
  const float* Wv             = (const float*)d_in[4];
  const float* Wo             = (const float*)d_in[5];
  const float* q_norm_w       = (const float*)d_in[6];
  const float* k_norm_w       = (const float*)d_in[7];
  const int*   ctx_pos        = (const int*)d_in[8];
  const int*   draft_pos      = (const int*)d_in[9];
  float*       out            = (float*)d_out;

  // workspace carve-out (all 256B aligned)
  char* ws = (char*)d_ws;
  size_t off = 0;
  auto alloc = [&](size_t bytes) -> void* {
    void* p = ws + off;
    off += (bytes + 255) & ~(size_t)255;
    return p;
  };
  u16* kvin   = (u16*)alloc((size_t)B_ * TOTAL_ * D_ * 2);        // [B,TOTAL,D]
  u16* wqb    = (u16*)alloc((size_t)H_  * HD_ * D_ * 2);
  u16* wkb    = (u16*)alloc((size_t)KVH_* HD_ * D_ * 2);
  u16* wvb    = (u16*)alloc((size_t)KVH_* HD_ * D_ * 2);
  u16* wob    = (u16*)alloc((size_t)D_  * H_ * HD_ * 2);
  u16* q_rope = (u16*)alloc((size_t)B_ * H_   * DRAFT_ * HD_ * 2);
  u16* k_rope = (u16*)alloc((size_t)B_ * KVH_ * TOTAL_ * HD_ * 2);
  u16* vT     = (u16*)alloc((size_t)B_ * KVH_ * HD_ * TOTAL_ * 2);
  u16* attnb  = (u16*)alloc((size_t)B_ * DRAFT_ * H_ * HD_ * 2);

  auto cgrid = [](long long n) {
    long long g = (n + 255) / 256;
    if (g > 8192) g = 8192;
    return (int)g;
  };

  // fp32 -> bf16 staging
  cvt_bf16_kernel<<<cgrid((long long)B_ * CTX_ * D_), 256, 0, stream>>>(
      context_hidden, kvin, B_, CTX_ * D_, (long long)TOTAL_ * D_);
  cvt_bf16_kernel<<<cgrid((long long)B_ * DRAFT_ * D_), 256, 0, stream>>>(
      draft_hidden, kvin + (size_t)CTX_ * D_, B_, DRAFT_ * D_, (long long)TOTAL_ * D_);
  cvt_bf16_kernel<<<cgrid((long long)H_ * HD_ * D_), 256, 0, stream>>>(
      Wq, wqb, 1, H_ * HD_ * D_, 0);
  cvt_bf16_kernel<<<cgrid((long long)KVH_ * HD_ * D_), 256, 0, stream>>>(
      Wk, wkb, 1, KVH_ * HD_ * D_, 0);
  cvt_bf16_kernel<<<cgrid((long long)KVH_ * HD_ * D_), 256, 0, stream>>>(
      Wv, wvb, 1, KVH_ * HD_ * D_, 0);
  cvt_bf16_kernel<<<cgrid((long long)D_ * H_ * HD_), 256, 0, stream>>>(
      Wo, wob, 1, D_ * H_ * HD_, 0);

  // projections (+norm+rope), layouts tuned for the attention WMMA fragments
  proj_kernel<<<128, 256, 0, stream>>>(kvin, wqb, q_norm_w, ctx_pos, draft_pos, q_rope, 0);
  proj_kernel<<<288, 256, 0, stream>>>(kvin, wkb, k_norm_w, ctx_pos, draft_pos, k_rope, 1);
  proj_kernel<<<288, 256, 0, stream>>>(kvin, wvb, q_norm_w, ctx_pos, draft_pos, vT,     2);

  // flash attention
  attn_kernel<<<B_ * KVH_ * 16, 256, 0, stream>>>(q_rope, k_rope, vT, attnb);

  // output projection -> fp32 d_out
  out_gemm_kernel<<<128, 256, 0, stream>>>(attnb, wob, out);
}
